// MultiFrameEstimatier_74586402062867
// MI455X (gfx1250) — matrix operand order repaired and checked by reading, not compiled, not run on hardware
//
#include <hip/hip_runtime.h>

typedef __attribute__((ext_vector_type(16))) _Float16 v16h;
typedef __attribute__((ext_vector_type(8)))  _Float16 v8h;
typedef __attribute__((ext_vector_type(8)))  float    v8f;
typedef __attribute__((ext_vector_type(4)))  float    v4f;
typedef __attribute__((ext_vector_type(4)))  unsigned int u32x4;
typedef __attribute__((ext_vector_type(4)))  int      i32x4;
typedef __attribute__((ext_vector_type(8)))  int      i32x8;

#define B_    2
#define N_    8192
#define M_    8192
#define D_    64
#define K_    16
#define C1_   128
#define C2_   128
#define CINP_ 96      // layer-1 K padded to 3 chunks of 32 (permuted: feats 0..63, xyz 64..66)

// ---------------------------------------------------------------------------
// Kernel 1: pre-swizzle weights to f16 in WMMA B-fragment order.
// Fragment order used here (16-bit B 32x16, A matches): lane = col N (mod 16), K = (lane>=16?16:0)+j.
// W1h[frag= tile*3+kc][lane][j] ; W1 input channels permuted: K<64 -> feats K,
// 64..66 -> xyz, 67..95 -> zero.  W2h[frag = tile*4+kc][lane][j].
// ---------------------------------------------------------------------------
__global__ void prep_weights_kernel(const float* __restrict__ W1,
                                    const float* __restrict__ W2,
                                    _Float16* __restrict__ W1h,
                                    _Float16* __restrict__ W2h) {
  int tid = blockIdx.x * blockDim.x + threadIdx.x;
  if (tid < 24 * 512) {                       // 8 tiles x 3 kc, 512 halves each
    int frag = tid >> 9;
    int lane = (tid >> 4) & 31;
    int j    = tid & 15;
    int tile = frag / 3, kc = frag % 3;
    int K    = kc * 32 + ((lane < 16) ? 0 : 16) + j;   // permuted input channel
    int ncol = tile * 16 + (lane & 15);
    float v = 0.f;
    if (K < 64)      v = W1[(3 + K) * C1_ + ncol];     // feature channels
    else if (K < 67) v = W1[(K - 64) * C1_ + ncol];    // xyz channels
    W1h[tid] = (_Float16)v;
  } else {
    int t = tid - 24 * 512;
    if (t < 32 * 512) {                       // 8 tiles x 4 kc
      int frag = t >> 9;
      int lane = (t >> 4) & 31;
      int j    = t & 15;
      int tile = frag >> 2, kc = frag & 3;
      int K    = kc * 32 + ((lane < 16) ? 0 : 16) + j;
      int ncol = tile * 16 + (lane & 15);
      W2h[t] = (_Float16)W2[K * C2_ + ncol];
    }
  }
}

// ---------------------------------------------------------------------------
// Kernel 2: brute-force KNN. xyz2 tiles staged into LDS by the Tensor Data
// Mover (TDM) when available (fallback: cooperative b128 copies). One thread
// per query, sorted top-16 kept in registers.
// ---------------------------------------------------------------------------
#define TILE_ 1024
__global__ void __launch_bounds__(256)
knn_kernel(const float* __restrict__ xyz1,
           const float* __restrict__ xyz2,
           int* __restrict__ idx_out) {
  __shared__ __align__(16) float st[TILE_ * 3];    // interleaved xyz tile (12 KB)
  int q = blockIdx.x * blockDim.x + threadIdx.x;   // 0 .. B*N-1 (block spans one b)
  int b = q / N_;
  float qx = xyz1[(size_t)q * 3 + 0];
  float qy = xyz1[(size_t)q * 3 + 1];
  float qz = xyz1[(size_t)q * 3 + 2];

  float bd[K_]; int bi[K_];
#pragma unroll
  for (int i = 0; i < K_; ++i) { bd[i] = 3.4e38f; bi[i] = 0; }

  const float* base2 = xyz2 + (size_t)b * M_ * 3;

  for (int t0 = 0; t0 < M_; t0 += TILE_) {
    __syncthreads();
#if __has_builtin(__builtin_amdgcn_tensor_load_to_lds)
    if (threadIdx.x < 32) {                        // wave 0 drives the TDM
      unsigned long long ga =
          (unsigned long long)(const void*)(base2 + (size_t)t0 * 3);
      unsigned la = (unsigned)(unsigned long long)(void*)st;
      u32x4 g0;
      g0[0] = 1u;                                  // count=1, user descriptor
      g0[1] = la;                                  // lds_addr
      g0[2] = (unsigned)ga;                        // global_addr[31:0]
      g0[3] = (unsigned)((ga >> 32) & 0x01FFFFFFu) | (2u << 30); // addr[56:32]|type=2
      i32x8 g1 = {};
      const unsigned TD0 = M_ * 3;                 // tensor_dim0 (4B units)
      const unsigned TL0 = TILE_ * 3;              // tile_dim0
      g1[0] = (int)(2u << 16);                     // data_size = 4 bytes
      g1[1] = (int)((TD0 & 0xFFFFu) << 16);        // tensor_dim0[15:0] -> bits 63:48
      g1[2] = (int)(((TD0 >> 16) & 0xFFFFu) | (1u << 16)); // dim0[31:16] | tensor_dim1=1
      g1[3] = (int)(TL0 << 16);                    // tile_dim0 -> bits 127:112
      g1[4] = 1;                                   // tile_dim1 = 1
      g1[5] = (int)TD0;                            // tensor_dim0_stride
      i32x4 z4 = {};
#if defined(__clang_major__) && (__clang_major__ >= 23)
      i32x8 z8 = {};
      __builtin_amdgcn_tensor_load_to_lds(g0, g1, z4, z4, z8, 0);
#else
      __builtin_amdgcn_tensor_load_to_lds(g0, g1, z4, z4, 0);
#endif
      __builtin_amdgcn_s_wait_tensorcnt(0);
    }
#else
    {
      const v4f* src = (const v4f*)(base2 + (size_t)t0 * 3);
      v4f* dst = (v4f*)st;
      for (int jj = threadIdx.x; jj < (TILE_ * 3) / 4; jj += blockDim.x)
        dst[jj] = src[jj];
    }
#endif
    __syncthreads();

    for (int j = 0; j < TILE_; ++j) {              // LDS broadcast reads
      float dx = st[j * 3 + 0] - qx;
      float dy = st[j * 3 + 1] - qy;
      float dz = st[j * 3 + 2] - qz;
      float d = dx * dx + dy * dy + dz * dz;
      if (d < bd[K_ - 1]) {                        // sorted insert
        bd[K_ - 1] = d; bi[K_ - 1] = t0 + j;
#pragma unroll
        for (int i = K_ - 1; i > 0; --i) {
          if (bd[i] < bd[i - 1]) {
            float td = bd[i]; bd[i] = bd[i - 1]; bd[i - 1] = td;
            int   ti = bi[i]; bi[i] = bi[i - 1]; bi[i - 1] = ti;
          }
        }
      }
    }
  }
#pragma unroll
  for (int i = 0; i < K_; ++i) idx_out[(size_t)q * K_ + i] = bi[i];
}

// ---------------------------------------------------------------------------
// Kernel 3: gather + 2-layer MLP via v_wmma_f32_16x16x32_f16 + max over K.
// One wave per query, 4 waves/block. A fragments read from row-major LDS as
// two contiguous b128 chunks per lane; B fragments are single 32-byte loads
// from the pre-swizzled weight arrays.
// ---------------------------------------------------------------------------
#define WPB_ 4

static __device__ __forceinline__ v16h load_a_frag(const _Float16* tile_base,
                                                   int Mrow, int kbase) {
  const v8h lo = *(const v8h*)(tile_base + Mrow * 128 + kbase);
  const v8h hi = *(const v8h*)(tile_base + Mrow * 128 + kbase + 8);
  return __builtin_shufflevector(lo, hi, 0, 1, 2, 3, 4, 5, 6, 7,
                                 8, 9, 10, 11, 12, 13, 14, 15);
}

__global__ void __launch_bounds__(32 * WPB_)
mlp_wmma_kernel(const float* __restrict__ xyz1,
                const float* __restrict__ xyz2,
                const float* __restrict__ feats2,
                const _Float16* __restrict__ W1h,
                const float* __restrict__ b1,
                const _Float16* __restrict__ W2h,
                const float* __restrict__ b2,
                const int* __restrict__ idx,
                float* __restrict__ out) {
  __shared__ __align__(16) _Float16 sA[WPB_][16 * 128];  // per-wave 16x128 tile

  int w    = threadIdx.x >> 5;
  int lane = threadIdx.x & 31;
  int q    = blockIdx.x * WPB_ + w;
  int b    = q / N_;

  __builtin_prefetch(W1h, 0, 1);            // global_prefetch_b8
  __builtin_prefetch(W2h, 0, 1);

  // ---- gather: permuted channels (feats 0..63, xyz 64..66, pad 67..95) ----
  const int* my = idx + (size_t)q * K_;
  int k = lane & 15, h = lane >> 4;
  int m = my[k];
  const float* f2 = feats2 + ((size_t)b * M_ + m) * D_ + h * 32;
  _Float16* row = &sA[w][k * 128];
#pragma unroll
  for (int c = 0; c < 4; ++c) {             // 8 floats -> packed 8 halves
    v4f x0 = *(const v4f*)(f2 + c * 8);
    v4f x1 = *(const v4f*)(f2 + c * 8 + 4);
    v8h hh;
#pragma unroll
    for (int e = 0; e < 4; ++e) { hh[e] = (_Float16)x0[e]; hh[4 + e] = (_Float16)x1[3 - 3 + e]; }
    *(v8h*)(row + h * 32 + c * 8) = hh;
  }
  if (h == 0) {                             // xyz diff -> cols 64..66, zeros 67..71
    const float* p2 = xyz2 + ((size_t)b * M_ + m) * 3;
    const float* p1 = xyz1 + (size_t)q * 3;
    v8h t = {};
    t[0] = (_Float16)(p2[0] - p1[0]);
    t[1] = (_Float16)(p2[1] - p1[1]);
    t[2] = (_Float16)(p2[2] - p1[2]);
    *(v8h*)(row + 64) = t;
  } else {                                  // zeros cols 72..95
    v8h z = {};
    *(v8h*)(row + 72) = z;
    *(v8h*)(row + 80) = z;
    *(v8h*)(row + 88) = z;
  }
  __syncthreads();

  int Mrow = lane & 15;
  int kbA  = (lane < 16) ? 0 : 16;          // A half-wave K offset
  int ncb  = lane & 15;

  // ---- layer 1 A fragments ----
  v16h a1[3];
#pragma unroll
  for (int kc = 0; kc < 3; ++kc) a1[kc] = load_a_frag(&sA[w][0], Mrow, kc * 32 + kbA);

  // ---- layer 1: h1 = leaky(A x W1 + b1) -> back to LDS as next A tile ----
#pragma unroll
  for (int tile = 0; tile < 8; ++tile) {
    int ncol = ncb + tile * 16;
    v8f acc = {};
#pragma unroll
    for (int kc = 0; kc < 3; ++kc) {
      v16h bf = *(const v16h*)(W1h + (((tile * 3 + kc) * 32 + lane) << 4));
      acc = __builtin_amdgcn_wmma_f32_16x16x32_f16(
          false, a1[kc], false, bf, (short)0, acc, false, false);
    }
    float bias = b1[ncol];
#pragma unroll
    for (int v = 0; v < 8; ++v) {
      float x = acc[v] + bias;
      x = (x > 0.f) ? x : 0.1f * x;
      int rrow = v + ((lane < 16) ? 0 : 8);  // D layout -> row-major LDS
      sA[w][rrow * 128 + ncol] = (_Float16)x;
    }
  }
  __syncthreads();

  // ---- layer 2 A fragments ----
  v16h a2[4];
#pragma unroll
  for (int kc = 0; kc < 4; ++kc) a2[kc] = load_a_frag(&sA[w][0], Mrow, kc * 32 + kbA);

  // ---- layer 2 + max over the 16 neighbor rows ----
  float* outp = out + (size_t)q * C2_;
#pragma unroll
  for (int tile = 0; tile < 8; ++tile) {
    int ncol = ncb + tile * 16;
    v8f acc = {};
#pragma unroll
    for (int kc = 0; kc < 4; ++kc) {
      v16h bf = *(const v16h*)(W2h + (((tile * 4 + kc) * 32 + lane) << 4));
      acc = __builtin_amdgcn_wmma_f32_16x16x32_f16(
          false, a2[kc], false, bf, (short)0, acc, false, false);
    }
    float bias = b2[ncol];
    float mx = -3.4e38f;
#pragma unroll
    for (int v = 0; v < 8; ++v) {
      float x = acc[v] + bias;
      x = (x > 0.f) ? x : 0.1f * x;
      mx = fmaxf(mx, x);
    }
    mx = fmaxf(mx, __shfl_xor(mx, 16, 32));
    if (lane < 16) outp[ncol] = mx;
  }
}

// ---------------------------------------------------------------------------
extern "C" void kernel_launch(void* const* d_in, const int* in_sizes, int n_in,
                              void* d_out, int out_size, void* d_ws, size_t ws_size,
                              hipStream_t stream) {
  (void)in_sizes; (void)n_in; (void)out_size; (void)ws_size;
  const float* xyz1   = (const float*)d_in[0];
  const float* xyz2   = (const float*)d_in[1];
  const float* feats2 = (const float*)d_in[2];
  const float* W1     = (const float*)d_in[3];
  const float* b1     = (const float*)d_in[4];
  const float* W2     = (const float*)d_in[5];
  const float* b2     = (const float*)d_in[6];
  // d_in[7] = nsample (16) -- compiled in as K_.

  char* ws = (char*)d_ws;
  int*      idx = (int*)ws;                                    // 1 MiB
  _Float16* W1h = (_Float16*)(ws + (size_t)B_ * N_ * K_ * sizeof(int));
  _Float16* W2h = W1h + 24 * 512;                              // 12288 halves
  float*    out = (float*)d_out;

  int wtotal = 24 * 512 + 32 * 512;
  prep_weights_kernel<<<(wtotal + 255) / 256, 256, 0, stream>>>(W1, W2, W1h, W2h);
  knn_kernel<<<(B_ * N_) / 256, 256, 0, stream>>>(xyz1, xyz2, idx);
  mlp_wmma_kernel<<<(B_ * N_) / WPB_, 32 * WPB_, 0, stream>>>(
      xyz1, xyz2, feats2, W1h, b1, W2h, b2, idx, out);
}
